// ScannedRNN_4810363373057
// MI455X (gfx1250) — compile-verified, matrix operand order checked by reading, not run on hardware
//
#include <hip/hip_runtime.h>
#include <hip/hip_bf16.h>

// ---------------------------------------------------------------------------
// GRU scanned RNN, T=512, B=256, H=512, on gfx1250 (MI455X).
// bf16 WMMA (v_wmma_f32_16x16x32_bf16) with f32 accumulation.
// - Weights packed to WMMA B-layout, staged ONCE into 96 KB LDS per block,
//   reused for all 512 timesteps (ds_load_b128 in the hot loop).
// - x converted to bf16 one step ahead inside the elementwise phase.
// - Persistent kernel, one atomic grid barrier per timestep.
// ---------------------------------------------------------------------------

typedef __bf16 bf16_t;
typedef __attribute__((ext_vector_type(16))) __bf16 v16bf;
typedef __attribute__((ext_vector_type(8)))  __bf16 v8bf;
typedef __attribute__((ext_vector_type(8)))  float   v8f;

#define TT   512
#define BB   256
#define HH   512
#define NT_N (HH/16)                               // 32 column tiles
#define NT_M (BB/16)                               // 16 row tiles
#define KC   (HH/32)                               // 16 K-chunks of 32
#define WPB  8                                     // waves per block
#define NBLK ((NT_M*NT_N)/WPB)                     // 64 blocks
#define MATSZ (HH*HH)                              // bf16 elems per packed matrix
#define PANEL (KC*512)                             // 8192 elems = one nt panel
#define LDS_BYTES (6*PANEL*2)                      // 96 KB

__device__ __forceinline__ float sigmoid_fast(float x) {
    return 1.0f / (1.0f + __expf(-x));
}
__device__ __forceinline__ float tanh_fast(float x) {
    float e = __expf(-2.0f * x);
    return (1.0f - e) / (1.0f + e);
}

// combine two 8-wide bf16 loads into one WMMA operand register block
__device__ __forceinline__ v16bf ld16(const bf16_t* p0, const bf16_t* p1) {
    v8bf a = *(const v8bf*)p0;
    v8bf b = *(const v8bf*)p1;
    v16bf r;
#pragma unroll
    for (int i = 0; i < 8; ++i) { r[i] = a[i]; r[8 + i] = b[i]; }
    return r;
}

// ---------------------------------------------------------------------------
// Init: h0 = resets[0] ? 0 : init_h (f32 + bf16), xbf[0] = bf16(x[0]), ctr=0.
// ---------------------------------------------------------------------------
__global__ void gru_init(const float* __restrict__ init_h,
                         const unsigned char* __restrict__ resets,
                         const float* __restrict__ x,
                         float* __restrict__ hbuf, bf16_t* __restrict__ hbf,
                         bf16_t* __restrict__ xbf,
                         unsigned* __restrict__ counter) {
    int idx = blockIdx.x * blockDim.x + threadIdx.x;
    if (idx == 0) *counter = 0u;
    if (idx < BB * HH) {
        int b = idx / HH;
        float v = resets[b] ? 0.0f : init_h[idx];
        hbuf[idx] = v;
        hbf[idx]  = (bf16_t)v;
        xbf[idx]  = (bf16_t)x[idx];
    }
}

// ---------------------------------------------------------------------------
// Pack six HxH f32 weight matrices into WMMA bf16 B-operand tile layout.
// B is 32x16 (KxN). Per tile: lane l holds 16 contiguous bf16:
//   n = nt*16 + (l&15);  kb = kc*32 + (l<16 ? 0 : 8)
//   elem e<8 -> k = kb+e ;  e>=8 -> k = kb+16+(e-8)
// ---------------------------------------------------------------------------
__global__ void gru_pack_w(const float* __restrict__ Wir, const float* __restrict__ Wiz,
                           const float* __restrict__ Win, const float* __restrict__ Whr,
                           const float* __restrict__ Whz, const float* __restrict__ Whn,
                           bf16_t* __restrict__ out) {
    long idx = (long)blockIdx.x * blockDim.x + threadIdx.x;
    if (idx >= 6L * MATSZ) return;
    int m    = (int)(idx / MATSZ);
    int r    = (int)(idx % MATSZ);
    int e    = r & 15;
    int lane = (r >> 4) & 31;
    int tile = r >> 9;                 // nt*KC + kc
    int kc   = tile % KC;
    int nt   = tile / KC;
    int n    = nt * 16 + (lane & 15);
    int kb   = kc * 32 + ((lane < 16) ? 0 : 8);
    int k    = kb + ((e < 8) ? e : (8 + e));   // e>=8 -> kb+16+(e-8)
    const float* W;
    switch (m) {
        case 0: W = Wir; break;
        case 1: W = Wiz; break;
        case 2: W = Win; break;
        case 3: W = Whr; break;
        case 4: W = Whz; break;
        default: W = Whn; break;
    }
    out[idx] = (bf16_t)W[(size_t)k * HH + n];
}

// ---------------------------------------------------------------------------
// Persistent scan kernel. Block owns one column tile nt (weights in LDS);
// its 8 waves own 8 row tiles. Each wave keeps all 3 gates of its h-tile.
// ---------------------------------------------------------------------------
__global__ void __launch_bounds__(256, 1)
gru_scan(const float* __restrict__ x, const unsigned char* __restrict__ resets,
         const float* __restrict__ b_ir, const float* __restrict__ b_iz,
         const float* __restrict__ b_in, const float* __restrict__ b_hn,
         const bf16_t* __restrict__ wpk,
         float* __restrict__ hbuf, bf16_t* __restrict__ hbf,
         bf16_t* __restrict__ xbf,
         float* __restrict__ out, unsigned* __restrict__ counter) {
    extern __shared__ bf16_t lds_w[];             // 6 * PANEL bf16 = 96 KB

    const int lane = threadIdx.x & 31;
    const int wave = threadIdx.x >> 5;
    const int wid  = blockIdx.x * WPB + wave;     // 0..511
    const int nt   = wid >> 4;                    // column tile (block-uniform)
    const int mt   = wid & (NT_M - 1);            // row tile
    const int l15  = lane & 15;
    const int hi   = lane >> 4;                   // 0 or 1
    const int arow = mt * 16 + l15;               // A row for this lane
    const int ncol = nt * 16 + l15;               // C column for this lane

    // ---- stage this nt's six weight panels (6 x 16 KB, contiguous) to LDS ----
    {
        const int tid = threadIdx.x;
#pragma unroll
        for (int j = 0; j < 24; ++j) {
            const int u   = tid + j * 256;        // 16-byte unit, 0..6143
            const int m   = u >> 10;              // matrix id
            const int rem = u & 1023;             // unit within panel
            const bf16_t* g = wpk + (size_t)m * MATSZ + (size_t)nt * PANEL + (size_t)rem * 8;
            *(v8bf*)(lds_w + (size_t)u * 8) = *(const v8bf*)g;   // global -> LDS
        }
        // probe-verified CDNA5 split-counter wait (TENSORcnt is 0 here; no-op)
        __builtin_amdgcn_s_wait_tensorcnt(0);
        __syncthreads();
    }

    // per-lane bias scalars (C layout fixes N = lane&15 for all 8 VGPRs)
    const float bir = b_ir[ncol];
    const float biz = b_iz[ncol];
    const float bin = b_in[ncol];
    const float bhn = b_hn[ncol];

    // per-gate LDS base for this lane (advance 512 elems per kc)
    const bf16_t* lw0 = lds_w + 0 * PANEL + lane * 16;   // W_ir
    const bf16_t* lw1 = lds_w + 1 * PANEL + lane * 16;   // W_iz
    const bf16_t* lw2 = lds_w + 2 * PANEL + lane * 16;   // W_in
    const bf16_t* lw3 = lds_w + 3 * PANEL + lane * 16;   // W_hr
    const bf16_t* lw4 = lds_w + 4 * PANEL + lane * 16;   // W_hz
    const bf16_t* lw5 = lds_w + 5 * PANEL + lane * 16;   // W_hn

    for (int t = 0; t < TT; ++t) {
        const float*  hcur = hbuf + (size_t)(t & 1) * BB * HH;
        const bf16_t* hrow = hbf + (size_t)(t & 1) * BB * HH + (size_t)arow * HH;
        const bf16_t* xrow = xbf + (size_t)(t & 1) * BB * HH + (size_t)arow * HH;
        float*  hnext = hbuf + (size_t)((t + 1) & 1) * BB * HH;
        bf16_t* hbfn  = hbf  + (size_t)((t + 1) & 1) * BB * HH;
        bf16_t* xbfn  = xbf  + (size_t)((t + 1) & 1) * BB * HH;

        // gfx1250 prefetch of next timestep's f32 x tile (global_prefetch_b8)
        if (t + 1 < TT)
            __builtin_prefetch(x + ((size_t)(t + 1) * BB + arow) * HH + ncol, 0, 1);

        v8f cr = {}, cz = {}, cnx = {}, cnh = {};

#pragma unroll 2
        for (int kc = 0; kc < KC; ++kc) {
            const int kb = kc * 32 + hi * 8;
            const int wo = kc * 512;
            // ---- A operands (both bf16, 2x16B loads each) ----
            v16bf ax = ld16(xrow + kb, xrow + kb + 16);
            v16bf ah = ld16(hrow + kb, hrow + kb + 16);
            // ---- B operands from LDS (ds_load_b128 pairs) ----
            v16bf bri = ld16(lw0 + wo, lw0 + wo + 8);
            v16bf brh = ld16(lw3 + wo, lw3 + wo + 8);
            v16bf bzi = ld16(lw1 + wo, lw1 + wo + 8);
            v16bf bzh = ld16(lw4 + wo, lw4 + wo + 8);
            v16bf bni = ld16(lw2 + wo, lw2 + wo + 8);
            v16bf bnh = ld16(lw5 + wo, lw5 + wo + 8);

            cr  = __builtin_amdgcn_wmma_f32_16x16x32_bf16(false, ax, false, bri, (short)0, cr,  false, false);
            cr  = __builtin_amdgcn_wmma_f32_16x16x32_bf16(false, ah, false, brh, (short)0, cr,  false, false);
            cz  = __builtin_amdgcn_wmma_f32_16x16x32_bf16(false, ax, false, bzi, (short)0, cz,  false, false);
            cz  = __builtin_amdgcn_wmma_f32_16x16x32_bf16(false, ah, false, bzh, (short)0, cz,  false, false);
            cnx = __builtin_amdgcn_wmma_f32_16x16x32_bf16(false, ax, false, bni, (short)0, cnx, false, false);
            cnh = __builtin_amdgcn_wmma_f32_16x16x32_bf16(false, ah, false, bnh, (short)0, cnh, false, false);
        }

        // ---- gates + blend + stores (row = mt*16 + hi*8 + i, col = ncol) ----
        // also convert x[t+1] tile to bf16 for the next step (double-buffered)
#pragma unroll
        for (int i = 0; i < 8; ++i) {
            const int row = mt * 16 + hi * 8 + i;
            const float hprev = hcur[(size_t)row * HH + ncol];
            const float r  = sigmoid_fast(cr[i] + bir);
            const float z  = sigmoid_fast(cz[i] + biz);
            const float nn = tanh_fast(cnx[i] + bin + r * (cnh[i] + bhn));
            const float hnew = (1.0f - z) * nn + z * hprev;
            out[((size_t)t * BB + row) * HH + ncol] = hnew;
            float hst = hnew;
            if (t + 1 < TT) {
                if (resets[(size_t)(t + 1) * BB + row]) hst = 0.0f;
                xbfn[(size_t)row * HH + ncol] =
                    (bf16_t)x[((size_t)(t + 1) * BB + row) * HH + ncol];
            }
            hnext[(size_t)row * HH + ncol] = hst;
            hbfn[(size_t)row * HH + ncol]  = (bf16_t)hst;
        }

        // ---- device-wide barrier (monotonic counter, reset per launch) ----
        __threadfence();
        __syncthreads();
        if (threadIdx.x == 0) {
            const unsigned tgt = (unsigned)(t + 1) * (unsigned)NBLK;
            atomicAdd(counter, 1u);
            while (__hip_atomic_load(counter, __ATOMIC_RELAXED,
                                     __HIP_MEMORY_SCOPE_AGENT) < tgt) {
                __builtin_amdgcn_s_sleep(1);
            }
        }
        __syncthreads();
        __threadfence();
    }
}

// ---------------------------------------------------------------------------
// Launch
// ---------------------------------------------------------------------------
extern "C" void kernel_launch(void* const* d_in, const int* in_sizes, int n_in,
                              void* d_out, int out_size, void* d_ws, size_t ws_size,
                              hipStream_t stream) {
    const float*         ins    = (const float*)d_in[0];
    const unsigned char* resets = (const unsigned char*)d_in[1];
    const float*         init_h = (const float*)d_in[2];
    const float* W_ir = (const float*)d_in[3];
    const float* W_iz = (const float*)d_in[4];
    const float* W_in = (const float*)d_in[5];
    const float* b_ir = (const float*)d_in[6];
    const float* b_iz = (const float*)d_in[7];
    const float* b_in = (const float*)d_in[8];
    const float* W_hr = (const float*)d_in[9];
    const float* W_hz = (const float*)d_in[10];
    const float* W_hn = (const float*)d_in[11];
    const float* b_hn = (const float*)d_in[12];
    float* out = (float*)d_out;

    char* ws = (char*)d_ws;
    unsigned* counter = (unsigned*)ws;                                   // 128 B
    float*  hbuf = (float*)(ws + 128);                                   // 2*B*H f32
    bf16_t* hbf  = (bf16_t*)(ws + 128 + 2L * BB * HH * 4);               // 2*B*H bf16
    bf16_t* xbf  = (bf16_t*)(ws + 128 + 2L * BB * HH * 4 + 2L * BB * HH * 2);
    bf16_t* wpk  = (bf16_t*)(ws + 128 + 2L * BB * HH * 4 + 4L * BB * HH * 2);

    (void)hipFuncSetAttribute((const void*)gru_scan,
                              hipFuncAttributeMaxDynamicSharedMemorySize,
                              LDS_BYTES);

    gru_init<<<(BB * HH + 255) / 256, 256, 0, stream>>>(init_h, resets, ins,
                                                        hbuf, hbf, xbf, counter);
    gru_pack_w<<<(int)((6L * MATSZ + 255) / 256), 256, 0, stream>>>(
        W_ir, W_iz, W_in, W_hr, W_hz, W_hn, wpk);
    gru_scan<<<NBLK, 256, LDS_BYTES, stream>>>(ins, resets, b_ir, b_iz, b_in, b_hn,
                                               wpk, hbuf, hbf, xbf, out, counter);
}
